// ComplexPolarAttention_69569880260836
// MI455X (gfx1250) — matrix operand order repaired and checked by reading, not compiled, not run on hardware
//
#include <hip/hip_runtime.h>
#include <hip/hip_bf16.h>

typedef _Float16 h16;
typedef __attribute__((ext_vector_type(16))) _Float16 v16h;
typedef __attribute__((ext_vector_type(8)))  _Float16 v8h;
typedef __attribute__((ext_vector_type(8)))  float    v8f;
typedef __attribute__((ext_vector_type(4)))  float    v4f;

#define N_NODES 8192
#define DMODEL  256
#define NEDGES  262144

// ---------------------------------------------------------------------------
// CDNA5 async copy: global -> LDS without a VGPR round trip (ASYNCcnt).
// Per CDNA5 ISA: each lane copies 16B from its global address to
// its LDS address; completion tracked with ASYNCcnt.
// ---------------------------------------------------------------------------
static __device__ inline void async_copy_b128(void* lds_dst, const void* gsrc) {
  unsigned dst = (unsigned)(unsigned long long)lds_dst;   // low 32b = LDS offset
  unsigned long long src = (unsigned long long)gsrc;
  asm volatile("global_load_async_to_lds_b128 %0, %1, off"
               :: "v"(dst), "v"(src) : "memory");
}

static __device__ inline void wait_async() {
  asm volatile("s_wait_asynccnt 0x0" ::: "memory");
}

static __device__ inline v16h make_v16(v8h lo, v8h hi) {
  union { v16h v; v8h h[2]; } u;
  u.h[0] = lo; u.h[1] = hi;
  return u.v;
}

// A fragment: 16x32 f16 tile, row-major with `stride` halves per row.
// Lane L holds row L%16; halves 0..7 = K[(L/16)*8 .. +7], 8..15 = K[16+(L/16)*8 ..].
static __device__ inline v16h load_a_frag(const h16* tile, int stride, int lane) {
  const int row = lane & 15;
  const int kk  = (lane >> 4) << 3;
  const h16* p  = tile + row * stride;
  v8h lo = *(const v8h*)(p + kk);
  v8h hi = *(const v8h*)(p + 16 + kk);
  return make_v16(lo, hi);
}

// B fragment: tile staged as [n][k] (stride halves in k).
// Lane L holds column L%16; halves 0..7 = K[(L/16)*8 .. +7], 8..15 = K[16+(L/16)*8 ..].
static __device__ inline v16h load_b_frag(const h16* tile, int stride, int lane) {
  const int col = lane & 15;
  const int kk  = (lane >> 4) << 3;
  const h16* p  = tile + col * stride + kk;
  v8h lo = *(const v8h*)(p);
  v8h hi = *(const v8h*)(p + 16);
  return make_v16(lo, hi);
}

static __device__ inline v8f wmma_f16(v16h a, v16h b, v8f c) {
  return __builtin_amdgcn_wmma_f32_16x16x32_f16(false, a, false, b, (short)0, c,
                                                false, false);
}

// ---------------------------------------------------------------------------
// 1) elementwise prep: mc = mag*cos(phase), ms = mag*sin(phase)  (f16 out)
// ---------------------------------------------------------------------------
__global__ __launch_bounds__(256) void prep_kernel(
    const float* __restrict__ mag, const float* __restrict__ phase,
    h16* __restrict__ mcH, h16* __restrict__ msH) {
  const size_t i = (size_t)blockIdx.x * blockDim.x + threadIdx.x;
  const float m = mag[i];
  float s, c;
  __sincosf(phase[i], &s, &c);
  mcH[i] = (h16)(m * c);
  msH[i] = (h16)(m * s);
}

// ---------------------------------------------------------------------------
// 2) tiled transpose: mag/phase [N][D] f32 -> magT/phaseT [D][N] f16
// ---------------------------------------------------------------------------
__global__ __launch_bounds__(256) void transpose_kernel(
    const float* __restrict__ mag, const float* __restrict__ phase,
    h16* __restrict__ magT, h16* __restrict__ phaseT) {
  __shared__ h16 tm[32][33];
  __shared__ h16 tp[32][33];
  const int cbase = blockIdx.x * 32;   // D dim
  const int rbase = blockIdx.y * 32;   // N dim
  const int t  = threadIdx.x;
  const int r  = t >> 3;
  const int c0 = (t & 7) << 2;
  const v4f vm = *(const v4f*)(mag   + (size_t)(rbase + r) * DMODEL + cbase + c0);
  const v4f vp = *(const v4f*)(phase + (size_t)(rbase + r) * DMODEL + cbase + c0);
#pragma unroll
  for (int i = 0; i < 4; ++i) { tm[r][c0 + i] = (h16)vm[i]; tp[r][c0 + i] = (h16)vp[i]; }
  __syncthreads();
  const int c  = t >> 3;
  const int r0 = (t & 7) << 2;
#pragma unroll
  for (int i = 0; i < 4; ++i) {
    magT[(size_t)(cbase + c) * N_NODES + rbase + r0 + i]   = tm[r0 + i][c];
    phaseT[(size_t)(cbase + c) * N_NODES + rbase + r0 + i] = tp[r0 + i][c];
  }
}

// ---------------------------------------------------------------------------
// 3) scores = (mc@mc^T + ms@ms^T)/16.  Block tile 64(i) x 128(j), 8 waves 4x2,
//    double-buffered LDS filled by async global->LDS copies.
//    Buffer layout (halves): Amc[0,2048) Ams[2048,4096) Bmc[4096,8192) Bms[8192,12288)
// ---------------------------------------------------------------------------
__global__ __launch_bounds__(256) void gemm_scores_kernel(
    const h16* __restrict__ mcH, const h16* __restrict__ msH,
    float* __restrict__ scores) {
  __shared__ __align__(32) h16 lds[2 * 12288];

  const int jbase = blockIdx.x * 128;
  const int ibase = blockIdx.y * 64;
  const int t    = threadIdx.x;
  const int lane = t & 31;
  const int wave = t >> 5;
  const int wm   = wave & 3;
  const int wn   = wave >> 2;

  const int arow = t >> 2, acol = (t & 3) << 3;   // A tiles: 64 x 32
  const int brow = t >> 1, bcol = (t & 1) << 4;   // B tiles: 128 x 32

  const h16* gAmc = mcH + (size_t)(ibase + arow) * DMODEL + acol;
  const h16* gAms = msH + (size_t)(ibase + arow) * DMODEL + acol;
  const h16* gBmc = mcH + (size_t)(jbase + brow) * DMODEL + bcol;
  const h16* gBms = msH + (size_t)(jbase + brow) * DMODEL + bcol;

  auto stage = [&](int buf, int k0) {
    h16* b = lds + buf * 12288;
    async_copy_b128(b + arow * 32 + acol,            gAmc + k0);
    async_copy_b128(b + 2048 + arow * 32 + acol,     gAms + k0);
    async_copy_b128(b + 4096 + brow * 32 + bcol,     gBmc + k0);
    async_copy_b128(b + 4096 + brow * 32 + bcol + 8, gBmc + k0 + 8);
    async_copy_b128(b + 8192 + brow * 32 + bcol,     gBms + k0);
    async_copy_b128(b + 8192 + brow * 32 + bcol + 8, gBms + k0 + 8);
  };

  v8f acc[4] = {};

  stage(0, 0);
  wait_async();
  __syncthreads();

  for (int it = 0; it < DMODEL / 32; ++it) {
    const h16* b = lds + (it & 1) * 12288;
    if (it + 1 < DMODEL / 32) stage((it + 1) & 1, (it + 1) * 32);

    // hoist all fragment loads so dscnt waits batch and WMMAs go back-to-back
    const v16h aMc = load_a_frag(b + (16 * wm) * 32, 32, lane);
    const v16h aMs = load_a_frag(b + 2048 + (16 * wm) * 32, 32, lane);
    v16h bMc[4], bMs[4];
#pragma unroll
    for (int bt = 0; bt < 4; ++bt) {
      bMc[bt] = load_b_frag(b + 4096 + (64 * wn + 16 * bt) * 32, 32, lane);
      bMs[bt] = load_b_frag(b + 8192 + (64 * wn + 16 * bt) * 32, 32, lane);
    }
#pragma unroll
    for (int bt = 0; bt < 4; ++bt) {
      acc[bt] = wmma_f16(aMc, bMc[bt], acc[bt]);
      acc[bt] = wmma_f16(aMs, bMs[bt], acc[bt]);
    }

    wait_async();
    __syncthreads();
  }

  const float scale = 1.0f / 16.0f;  // 1/sqrt(256)
  const int col = lane & 15;
  const int g   = lane >> 4;
  const int r0  = ibase + 16 * wm + 8 * g;
#pragma unroll
  for (int bt = 0; bt < 4; ++bt) {
    const int c0 = jbase + 64 * wn + 16 * bt + col;
#pragma unroll
    for (int v = 0; v < 8; ++v)
      scores[(size_t)(r0 + v) * N_NODES + c0] = acc[bt][v] * scale;
  }
}

// ---------------------------------------------------------------------------
// 4) per-edge MLP bias scatter-added onto scores
// ---------------------------------------------------------------------------
__global__ __launch_bounds__(256) void edge_bias_kernel(
    const float* __restrict__ edge_attr, const long long* __restrict__ edge_index,
    const float* __restrict__ w1, const float* __restrict__ b1,
    const float* __restrict__ w2, const float* __restrict__ b2,
    const float* __restrict__ dist_scale, float* __restrict__ scores) {
  const int e = blockIdx.x * blockDim.x + threadIdx.x;
  if (e >= NEDGES) return;
  const float a = edge_attr[e];
  float acc = 0.f;
#pragma unroll
  for (int k = 0; k < 64; ++k) {
    const float x   = a * w1[k] + b1[k];
    const float sig = 1.f / (1.f + __expf(-x));
    acc += (x * sig) * w2[k];
  }
  const float bias = acc + b2[0] + dist_scale[0] * a;
  const long long i = edge_index[e];
  const long long j = edge_index[NEDGES + e];
  atomicAdd(scores + (size_t)i * N_NODES + j, bias);
}

// ---------------------------------------------------------------------------
// 5) row softmax (fp32 in, f16 attn out)
// ---------------------------------------------------------------------------
__global__ __launch_bounds__(256) void softmax_kernel(
    const float* __restrict__ scores, h16* __restrict__ attnH) {
  __shared__ float red[256];
  const int r = blockIdx.x;
  const int t = threadIdx.x;
  const float* row = scores + (size_t)r * N_NODES;

  float mx = -3.4e38f;
  for (int c = t; c < N_NODES; c += 256) mx = fmaxf(mx, row[c]);
  red[t] = mx; __syncthreads();
  for (int s = 128; s > 0; s >>= 1) {
    if (t < s) red[t] = fmaxf(red[t], red[t + s]);
    __syncthreads();
  }
  mx = red[0]; __syncthreads();

  float sum = 0.f;
  for (int c = t; c < N_NODES; c += 256) sum += __expf(row[c] - mx);
  red[t] = sum; __syncthreads();
  for (int s = 128; s > 0; s >>= 1) {
    if (t < s) red[t] += red[t + s];
    __syncthreads();
  }
  const float inv = 1.f / red[0];

  h16* orow = attnH + (size_t)r * N_NODES;
  for (int c = t; c < N_NODES; c += 256)
    orow[c] = (h16)(__expf(row[c] - mx) * inv);
}

// ---------------------------------------------------------------------------
// 6) new_mag = attn@mag, new_phase = attn@phase.  Block tile 64x64, K=8192.
//    magT/phaseT are pre-transposed so every tile is a contiguous async copy.
//    Buffer layout (halves): A[0,2048) Bm[2048,4096) Bp[4096,6144)
// ---------------------------------------------------------------------------
__global__ __launch_bounds__(256) void gemm_out_kernel(
    const h16* __restrict__ attnH, const h16* __restrict__ magT,
    const h16* __restrict__ phaseT, float* __restrict__ outMag,
    float* __restrict__ outPhase) {
  __shared__ __align__(32) h16 lds[2 * 6144];

  const int cbase = blockIdx.x * 64;
  const int rbase = blockIdx.y * 64;
  const int t    = threadIdx.x;
  const int lane = t & 31;
  const int wave = t >> 5;
  const int wm   = wave & 3;
  const int wn   = wave >> 2;

  const int arow = t >> 2, acol = (t & 3) << 3;   // 64 x 32 tiles

  const h16* gA  = attnH  + (size_t)(rbase + arow) * N_NODES + acol;
  const h16* gBm = magT   + (size_t)(cbase + arow) * N_NODES + acol;
  const h16* gBp = phaseT + (size_t)(cbase + arow) * N_NODES + acol;

  auto stage = [&](int buf, int k0) {
    h16* b = lds + buf * 6144;
    async_copy_b128(b + arow * 32 + acol,        gA + k0);
    async_copy_b128(b + 2048 + arow * 32 + acol, gBm + k0);
    async_copy_b128(b + 4096 + arow * 32 + acol, gBp + k0);
  };

  v8f accM[2] = {};
  v8f accP[2] = {};

  stage(0, 0);
  wait_async();
  __syncthreads();

  for (int it = 0; it < N_NODES / 32; ++it) {
    const h16* b = lds + (it & 1) * 6144;
    if (it + 1 < N_NODES / 32) stage((it + 1) & 1, (it + 1) * 32);

    const v16h a = load_a_frag(b + (16 * wm) * 32, 32, lane);
    v16h bm[2], bp[2];
#pragma unroll
    for (int bt = 0; bt < 2; ++bt) {
      bm[bt] = load_b_frag(b + 2048 + (32 * wn + 16 * bt) * 32, 32, lane);
      bp[bt] = load_b_frag(b + 4096 + (32 * wn + 16 * bt) * 32, 32, lane);
    }
#pragma unroll
    for (int bt = 0; bt < 2; ++bt) {
      accM[bt] = wmma_f16(a, bm[bt], accM[bt]);
      accP[bt] = wmma_f16(a, bp[bt], accP[bt]);
    }

    wait_async();
    __syncthreads();
  }

  const int col = lane & 15;
  const int g   = lane >> 4;
  const int r0  = rbase + 16 * wm + 8 * g;
#pragma unroll
  for (int bt = 0; bt < 2; ++bt) {
    const int c0 = cbase + 32 * wn + 16 * bt + col;
#pragma unroll
    for (int v = 0; v < 8; ++v) {
      outMag[(size_t)(r0 + v) * DMODEL + c0]   = accM[bt][v];
      outPhase[(size_t)(r0 + v) * DMODEL + c0] = accP[bt][v];
    }
  }
}

// ---------------------------------------------------------------------------
extern "C" void kernel_launch(void* const* d_in, const int* in_sizes, int n_in,
                              void* d_out, int out_size, void* d_ws, size_t ws_size,
                              hipStream_t stream) {
  const float*     mag        = (const float*)d_in[0];
  const float*     phase      = (const float*)d_in[1];
  const float*     edge_attr  = (const float*)d_in[2];
  const long long* edge_index = (const long long*)d_in[3];
  const float*     w1         = (const float*)d_in[4];
  const float*     b1         = (const float*)d_in[5];
  const float*     w2         = (const float*)d_in[6];
  const float*     b2         = (const float*)d_in[7];
  const float*     dist_scale = (const float*)d_in[8];

  const size_t nd = (size_t)N_NODES * DMODEL;
  const size_t nn = (size_t)N_NODES * N_NODES;

  char* ws = (char*)d_ws;
  h16* mcH    = (h16*)ws;  ws += nd * sizeof(h16);
  h16* msH    = (h16*)ws;  ws += nd * sizeof(h16);
  h16* magT   = (h16*)ws;  ws += nd * sizeof(h16);
  h16* phaseT = (h16*)ws;  ws += nd * sizeof(h16);
  float* scores = (float*)ws;  ws += nn * sizeof(float);
  h16* attnH  = (h16*)ws;

  float* outMag   = (float*)d_out;
  float* outPhase = outMag + nd;

  prep_kernel<<<(int)(nd / 256), 256, 0, stream>>>(mag, phase, mcH, msH);
  transpose_kernel<<<dim3(DMODEL / 32, N_NODES / 32), 256, 0, stream>>>(
      mag, phase, magT, phaseT);
  gemm_scores_kernel<<<dim3(N_NODES / 128, N_NODES / 64), 256, 0, stream>>>(
      mcH, msH, scores);
  edge_bias_kernel<<<NEDGES / 256, 256, 0, stream>>>(edge_attr, edge_index, w1, b1,
                                                     w2, b2, dist_scale, scores);
  softmax_kernel<<<N_NODES, 256, 0, stream>>>(scores, attnH);
  gemm_out_kernel<<<dim3(DMODEL / 64, N_NODES / 64), 256, 0, stream>>>(
      attnH, magT, phaseT, outMag, outPhase);
}
